// MultiHeadAttention_54211077210696
// MI455X (gfx1250) — matrix operand order repaired and checked by reading, not compiled, not run on hardware
//
#include <hip/hip_runtime.h>

// ---------------------------------------------------------------------------
// MHA + residual + LayerNorm for MI455X (gfx1250, wave32, WMMA f16->f32).
// B=2, S=2048, D_MODEL=1024, H=16, Dk=Dv=64.  M = B*S = 4096.
// GEMMs register-blocked 16x64 per wave and software-pipelined: fragments for
// K-step kk+32 are loaded before the WMMAs of step kk execute, forcing
// overlapping lifetimes (distinct VGPRs) so loads run under the WMMA group.
// ---------------------------------------------------------------------------

typedef __attribute__((ext_vector_type(16))) _Float16 v16h;
typedef __attribute__((ext_vector_type(8)))  _Float16 v8h;
typedef __attribute__((ext_vector_type(8)))  float    v8f;
typedef __attribute__((ext_vector_type(4)))  float    v4f;

__device__ __forceinline__ v8f wmma_f16(v16h a, v16h b, v8f c) {
  // v_wmma_f32_16x16x32_f16  (neg_a, A, neg_b, B, c_mod, C, reuse_a, reuse_b)
  return __builtin_amdgcn_wmma_f32_16x16x32_f16(false, a, false, b, (short)0, c,
                                                false, false);
}

// B-fragment: 16 contiguous halves starting at p (k = hlf*16 + e).
__device__ __forceinline__ v16h load_h16(const _Float16* p) {
  v8h lo = *(const v8h*)p;
  v8h hi = *(const v8h*)(p + 8);
  v16h b;
#pragma unroll
  for (int j = 0; j < 8; ++j) { b[j] = lo[j]; b[8 + j] = hi[j]; }
  return b;
}

// A-fragment from f16 row-major: p = row_base + kk + hlf*16 ; runs at +0, +8.
__device__ __forceinline__ v16h load_a16(const _Float16* p) {
  v8h lo = *(const v8h*)p;
  v8h hi = *(const v8h*)(p + 8);
  v16h a;
#pragma unroll
  for (int j = 0; j < 8; ++j) { a[j] = lo[j]; a[8 + j] = hi[j]; }
  return a;
}

// A-fragment from f32 row-major with convert: p = row_base + kk + hlf*16.
__device__ __forceinline__ v16h load_a32(const float* p) {
  v4f x0 = *(const v4f*)p;
  v4f x1 = *(const v4f*)(p + 4);
  v4f y0 = *(const v4f*)(p + 8);
  v4f y1 = *(const v4f*)(p + 12);
  v16h a;
#pragma unroll
  for (int j = 0; j < 4; ++j) {
    a[j]      = (_Float16)x0[j];
    a[4 + j]  = (_Float16)x1[j];
    a[8 + j]  = (_Float16)y0[j];
    a[12 + j] = (_Float16)y1[j];
  }
  return a;
}

// ---------------------------------------------------------------------------
// Weight convert+transpose via LDS tile: W[k][n] f32 -> Wt[n][k] f16.
// block (32,8), grid (32,32); 32x32 tile, coalesced both ways.
// ---------------------------------------------------------------------------
__global__ __launch_bounds__(256) void wcvt_kernel(const float* __restrict__ W,
                                                   _Float16* __restrict__ Wt) {
  __shared__ _Float16 tile[32][33];
  const int tx = threadIdx.x, ty = threadIdx.y;
  const int nb = blockIdx.x * 32;   // n base
  const int kb = blockIdx.y * 32;   // k base
#pragma unroll
  for (int i = 0; i < 32; i += 8)
    tile[ty + i][tx] = (_Float16)W[(kb + ty + i) * 1024 + nb + tx];
  __syncthreads();
#pragma unroll
  for (int i = 0; i < 32; i += 8)
    Wt[(nb + ty + i) * 1024 + kb + tx] = tile[tx][ty + i];
}

// ---------------------------------------------------------------------------
// Projection GEMM: Out_f16[4096x1024] = X_f32 @ Wt^T (Wt[n][k] f16).
// One wave per 16x64 strip, double-buffered over the K loop.
// vstore=1 stores transposed (V -> Vt[dim][seq]).
// ---------------------------------------------------------------------------
__global__ __launch_bounds__(128) void proj_gemm_kernel(
    const float* __restrict__ X, const _Float16* __restrict__ Wt,
    _Float16* __restrict__ Out, int vstore) {
  const int lane = threadIdx.x & 31;
  const int strip = blockIdx.x * 4 + (threadIdx.x >> 5);  // 4096 strips
  const int tileM = strip >> 4;                           // 0..255
  const int nb = (strip & 15) * 64;                       // col base, 0..960
  const int mn = lane & 15;
  const int hlf = lane >> 4;

  const float* arow = X + (tileM * 16 + mn) * 1024;
  const _Float16* brow[4];
#pragma unroll
  for (int t = 0; t < 4; ++t) brow[t] = Wt + (nb + t * 16 + mn) * 1024;

  v8f acc[4] = {{}, {}, {}, {}};
  // Prologue: fragments for kk = 0.
  v16h a_cur = load_a32(arow + hlf * 16);
  v16h b_cur[4];
#pragma unroll
  for (int t = 0; t < 4; ++t) b_cur[t] = load_h16(brow[t] + hlf * 16);

  for (int kk = 0; kk < 1024; kk += 32) {
    const int kn = (kk + 32 < 1024) ? kk + 32 : kk;   // clamped next step
    if (kk + 64 < 1024) {
      __builtin_prefetch(arow + kk + 64 + hlf * 8, 0, 3);  // global_prefetch_b8
      __builtin_prefetch(brow[hlf * 2] + kk + 64, 0, 3);
    }
    // Issue next-step loads first; consumers are one iteration away.
    v16h a_nxt = load_a32(arow + kn + hlf * 16);
    v16h b_nxt[4];
#pragma unroll
    for (int t = 0; t < 4; ++t) b_nxt[t] = load_h16(brow[t] + kn + hlf * 16);
    // Current-step WMMAs overlap the in-flight loads above.
#pragma unroll
    for (int t = 0; t < 4; ++t) acc[t] = wmma_f16(a_cur, b_cur[t], acc[t]);
    a_cur = a_nxt;
#pragma unroll
    for (int t = 0; t < 4; ++t) b_cur[t] = b_nxt[t];
  }
#pragma unroll
  for (int t = 0; t < 4; ++t) {
    const int col = nb + t * 16 + mn;
#pragma unroll
    for (int r = 0; r < 8; ++r) {
      const int row = tileM * 16 + r + hlf * 8;
      if (vstore) Out[col * 4096 + row] = (_Float16)acc[t][r];
      else        Out[row * 1024 + col] = (_Float16)acc[t][r];
    }
  }
}

// ---------------------------------------------------------------------------
// Flash attention: 1 wave per 16 query rows of one (b,h).
// Scores live in WMMA accumulators; online softmax keeps per-lane partial
// row-sums (reduced once at the end); P transposed through 1KB LDS.
// K-frags are double-buffered across key-steps and V-frags load before the
// softmax VALU block, so global latency hides under exp/shuffle work.
// ---------------------------------------------------------------------------
__global__ __launch_bounds__(32) void attn_kernel(
    const _Float16* __restrict__ Qh, const _Float16* __restrict__ Kh,
    const _Float16* __restrict__ Vt, _Float16* __restrict__ Ctx) {
  __shared__ __align__(32) _Float16 pTile[16 * 32];

  const int lane = threadIdx.x & 31;
  const int mn = lane & 15;
  const int hlf = lane >> 4;
  const int qb = blockIdx.x;            // 0..127 query blocks
  const int bh = blockIdx.y;            // 0..31
  const int b = bh >> 4, h = bh & 15;
  const int rowbase = b * 2048 + qb * 16;
  const int cbase = h * 64;

  // Q A-frags for head-dim K-steps {0..31},{32..63}
  const v16h aq0 = load_a16(Qh + (rowbase + mn) * 1024 + cbase + 0  + hlf * 16);
  const v16h aq1 = load_a16(Qh + (rowbase + mn) * 1024 + cbase + 32 + hlf * 16);

  float m_i[8], l_i[8];
#pragma unroll
  for (int r = 0; r < 8; ++r) { m_i[r] = -3.0e38f; l_i[r] = 0.0f; }
  v8f o0 = {}, o1 = {}, o2 = {}, o3 = {};
  const v8f zero = {};

  // Prologue: K B-frags for kk = 0.
  v16h bk[4];
  {
    const _Float16* kp0 = Kh + (b * 2048 + mn) * 1024 + cbase;
    const _Float16* kp1 = Kh + (b * 2048 + 16 + mn) * 1024 + cbase;
    bk[0] = load_h16(kp0 + hlf * 16);
    bk[1] = load_h16(kp0 + 32 + hlf * 16);
    bk[2] = load_h16(kp1 + hlf * 16);
    bk[3] = load_h16(kp1 + 32 + hlf * 16);
  }

  for (int kk = 0; kk < 2048; kk += 32) {
    const int kr0 = b * 2048 + kk;
    const int knext = (kk + 32 < 2048) ? kk + 32 : kk;   // clamped
    const _Float16* vp = Vt + (cbase + mn) * 4096 + kr0 + hlf * 16;
    if (kk + 64 < 2048) {
      __builtin_prefetch(Kh + (kr0 + 64 + mn) * 1024 + cbase, 0, 3);
      __builtin_prefetch(vp + 64, 0, 3);
    }
    // Score WMMAs with current K fragments.
    v8f s0 = wmma_f16(aq0, bk[0], zero);
    s0     = wmma_f16(aq1, bk[1], s0);
    v8f s1 = wmma_f16(aq0, bk[2], zero);
    s1     = wmma_f16(aq1, bk[3], s1);

    // Issue V loads (needed after softmax) and next-step K loads (needed
    // next iteration); both hide under the softmax VALU block below.
    v16h bv[4];
    bv[0] = load_h16(vp);
    bv[1] = load_h16(vp + 16 * 4096);
    bv[2] = load_h16(vp + 32 * 4096);
    bv[3] = load_h16(vp + 48 * 4096);
    {
      const _Float16* np0 = Kh + (b * 2048 + knext + mn) * 1024 + cbase;
      const _Float16* np1 = Kh + (b * 2048 + knext + 16 + mn) * 1024 + cbase;
      bk[0] = load_h16(np0 + hlf * 16);
      bk[1] = load_h16(np0 + 32 + hlf * 16);
      bk[2] = load_h16(np1 + hlf * 16);
      bk[3] = load_h16(np1 + 32 + hlf * 16);
    }

#pragma unroll
    for (int r = 0; r < 8; ++r) {
      const float a0 = s0[r] * 0.125f;        // 1/sqrt(64)
      const float a1 = s1[r] * 0.125f;
      float cand = fmaxf(a0, a1);
#pragma unroll
      for (int off = 1; off < 16; off <<= 1)
        cand = fmaxf(cand, __shfl_xor(cand, off, 32));   // row max (uniform)
      const float newm = fmaxf(m_i[r], cand);
      const float alpha = __expf(m_i[r] - newm);
      const float p0 = __expf(a0 - newm);
      const float p1 = __expf(a1 - newm);
      l_i[r] = l_i[r] * alpha + (p0 + p1);    // per-lane partial row-sum
      m_i[r] = newm;
      o0[r] *= alpha; o1[r] *= alpha; o2[r] *= alpha; o3[r] *= alpha;
      pTile[(r + hlf * 8) * 32 + mn]      = (_Float16)p0;   // C-frag -> row major
      pTile[(r + hlf * 8) * 32 + 16 + mn] = (_Float16)p1;
    }
    __syncthreads();
    // P A-frag (16 queries x 32 keys) read back in A layout from LDS.
    const v16h ap = load_a16(&pTile[mn * 32 + hlf * 16]);
    o0 = wmma_f16(ap, bv[0], o0);
    o1 = wmma_f16(ap, bv[1], o1);
    o2 = wmma_f16(ap, bv[2], o2);
    o3 = wmma_f16(ap, bv[3], o3);
    __syncthreads();
  }

#pragma unroll
  for (int r = 0; r < 8; ++r) {
    float ls = l_i[r];
#pragma unroll
    for (int off = 1; off < 16; off <<= 1) ls += __shfl_xor(ls, off, 32);
    const float inv = 1.0f / ls;
    _Float16* cp = Ctx + (rowbase + r + hlf * 8) * 1024 + cbase + mn;
    cp[0]  = (_Float16)(o0[r] * inv);
    cp[16] = (_Float16)(o1[r] * inv);
    cp[32] = (_Float16)(o2[r] * inv);
    cp[48] = (_Float16)(o3[r] * inv);
  }
}

// ---------------------------------------------------------------------------
// Output GEMM + residual: Pre_f32[4096x1024] = Ctx_f16 @ WO (WOt[n][k]) + Xq
// Register-blocked 16x64 per wave, double-buffered like proj_gemm.
// ---------------------------------------------------------------------------
__global__ __launch_bounds__(128) void out_gemm_kernel(
    const _Float16* __restrict__ Ctx, const _Float16* __restrict__ WOt,
    const float* __restrict__ resid, float* __restrict__ Out) {
  const int lane = threadIdx.x & 31;
  const int strip = blockIdx.x * 4 + (threadIdx.x >> 5);
  const int tileM = strip >> 4;
  const int nb = (strip & 15) * 64;
  const int mn = lane & 15;
  const int hlf = lane >> 4;

  const _Float16* arow = Ctx + (tileM * 16 + mn) * 1024;
  const _Float16* brow[4];
#pragma unroll
  for (int t = 0; t < 4; ++t) brow[t] = WOt + (nb + t * 16 + mn) * 1024;

  v8f acc[4] = {{}, {}, {}, {}};
  v16h a_cur = load_a16(arow + hlf * 16);
  v16h b_cur[4];
#pragma unroll
  for (int t = 0; t < 4; ++t) b_cur[t] = load_h16(brow[t] + hlf * 16);

  for (int kk = 0; kk < 1024; kk += 32) {
    const int kn = (kk + 32 < 1024) ? kk + 32 : kk;
    if (kk + 64 < 1024) {
      __builtin_prefetch(arow + kk + 64 + hlf * 8, 0, 3);
      __builtin_prefetch(brow[hlf * 2] + kk + 64, 0, 3);
    }
    v16h a_nxt = load_a16(arow + kn + hlf * 16);
    v16h b_nxt[4];
#pragma unroll
    for (int t = 0; t < 4; ++t) b_nxt[t] = load_h16(brow[t] + kn + hlf * 16);
#pragma unroll
    for (int t = 0; t < 4; ++t) acc[t] = wmma_f16(a_cur, b_cur[t], acc[t]);
    a_cur = a_nxt;
#pragma unroll
    for (int t = 0; t < 4; ++t) b_cur[t] = b_nxt[t];
  }
#pragma unroll
  for (int t = 0; t < 4; ++t) {
    const int col = nb + t * 16 + mn;
#pragma unroll
    for (int r = 0; r < 8; ++r) {
      const int row = tileM * 16 + r + hlf * 8;
      Out[row * 1024 + col] = acc[t][r] + resid[row * 1024 + col];
    }
  }
}

// ---------------------------------------------------------------------------
// Row LayerNorm over 1024 (gamma=1, beta=0, eps=1e-5)
// ---------------------------------------------------------------------------
__global__ __launch_bounds__(256) void ln_kernel(const float* __restrict__ in,
                                                 float* __restrict__ out) {
  __shared__ float red[256];
  const int row = blockIdx.x;
  const int t = threadIdx.x;
  const float* p = in + row * 1024;
  float v0 = p[t], v1 = p[t + 256], v2 = p[t + 512], v3 = p[t + 768];
  red[t] = v0 + v1 + v2 + v3;
  __syncthreads();
  for (int off = 128; off > 0; off >>= 1) {
    if (t < off) red[t] += red[t + off];
    __syncthreads();
  }
  const float mean = red[0] * (1.0f / 1024.0f);
  __syncthreads();
  const float d0 = v0 - mean, d1 = v1 - mean, d2 = v2 - mean, d3 = v3 - mean;
  red[t] = d0 * d0 + d1 * d1 + d2 * d2 + d3 * d3;
  __syncthreads();
  for (int off = 128; off > 0; off >>= 1) {
    if (t < off) red[t] += red[t + off];
    __syncthreads();
  }
  const float inv = rsqrtf(red[0] * (1.0f / 1024.0f) + 1e-5f);
  float* q = out + row * 1024;
  q[t] = d0 * inv;
  q[t + 256] = d1 * inv;
  q[t + 512] = d2 * inv;
  q[t + 768] = d3 * inv;
}

// ---------------------------------------------------------------------------
extern "C" void kernel_launch(void* const* d_in, const int* in_sizes, int n_in,
                              void* d_out, int out_size, void* d_ws,
                              size_t ws_size, hipStream_t stream) {
  (void)in_sizes; (void)n_in; (void)out_size; (void)ws_size;
  const float* Xq = (const float*)d_in[0];
  const float* Xk = (const float*)d_in[1];
  const float* Xv = (const float*)d_in[2];
  const float* WQ = (const float*)d_in[3];
  const float* WK = (const float*)d_in[4];
  const float* WV = (const float*)d_in[5];
  const float* WO = (const float*)d_in[6];
  float* out = (float*)d_out;

  char* ws = (char*)d_ws;
  const size_t MB = 1024 * 1024;
  _Float16* WQt = (_Float16*)(ws + 0 * MB);   // 2 MB each
  _Float16* WKt = (_Float16*)(ws + 2 * MB);
  _Float16* WVt = (_Float16*)(ws + 4 * MB);
  _Float16* WOt = (_Float16*)(ws + 6 * MB);
  _Float16* Qh  = (_Float16*)(ws + 8 * MB);   // [4096][1024] f16
  _Float16* Kh  = (_Float16*)(ws + 16 * MB);  // [4096][1024] f16
  _Float16* Vt  = (_Float16*)(ws + 24 * MB);  // [1024][4096] f16 (transposed)
  _Float16* Ctx = (_Float16*)(ws + 32 * MB);  // [4096][1024] f16
  float*    Pre = (float*)(ws + 8 * MB);      // [4096][1024] f32, aliases Qh/Kh

  dim3 tblk(32, 8), tgrid(32, 32);
  wcvt_kernel<<<tgrid, tblk, 0, stream>>>(WQ, WQt);
  wcvt_kernel<<<tgrid, tblk, 0, stream>>>(WK, WKt);
  wcvt_kernel<<<tgrid, tblk, 0, stream>>>(WV, WVt);
  wcvt_kernel<<<tgrid, tblk, 0, stream>>>(WO, WOt);

  proj_gemm_kernel<<<1024, 128, 0, stream>>>(Xq, WQt, Qh, 0);
  proj_gemm_kernel<<<1024, 128, 0, stream>>>(Xk, WKt, Kh, 0);
  proj_gemm_kernel<<<1024, 128, 0, stream>>>(Xv, WVt, Vt, 1);

  attn_kernel<<<dim3(128, 32), 32, 0, stream>>>(Qh, Kh, Vt, Ctx);

  out_gemm_kernel<<<1024, 128, 0, stream>>>(Ctx, WOt, Xq, Pre);
  ln_kernel<<<4096, 256, 0, stream>>>(Pre, out);
}
